// MultiHeadAttention_2010044694646
// MI455X (gfx1250) — compile-verified, matrix operand order checked by reading, not run on hardware
//
#include <hip/hip_runtime.h>
#include <stdint.h>

// ---------------------------------------------------------------------------
// MHA forward for MI455X (gfx1250, wave32, WMMA bf16 16x16x32).
//   B=2, S=Q=KV=2048, D_MODEL=1024, H=16, d=64
// Workspace layout (bytes), total 40 MB:
//   [ 0MB) WqT  bf16 [1024 N][1024 K]
//   [ 2MB) WkT  bf16
//   [ 4MB) WvT  bf16
//   [ 6MB) WoT  bf16
//   [ 8MB) Qb   bf16 (b,h,s,64)
//   [16MB) Kb   bf16 (b,h,s,64)
//   [24MB) VTb  bf16 (b,h,64,s)   (pre-transposed for P@V B-operand)
//   [32MB) CTX  bf16 (b,s,h*64)
// ---------------------------------------------------------------------------

#define BATCH 2
#define SEQ   2048
#define DM    1024
#define NH    16
#define DH    64

typedef __attribute__((ext_vector_type(16))) __bf16 v16bf;
typedef __attribute__((ext_vector_type(2)))  __bf16 v2bf;
typedef __attribute__((ext_vector_type(8)))  float  v8f;

// Scalar fp32 -> bf16 (RNE) via native fptrunc.
static __device__ __forceinline__ __bf16 f2bf(float f) { return (__bf16)f; }

// Packed pair conversion: one v_cvt_pk_bf16_f32 where the builtin exists.
static __device__ __forceinline__ v2bf f2bf2(float a, float b) {
#if __has_builtin(__builtin_amdgcn_cvt_pk_bf16_f32)
  return __builtin_amdgcn_cvt_pk_bf16_f32(a, b);
#else
  v2bf r; r[0] = (__bf16)a; r[1] = (__bf16)b; return r;
#endif
}

static __device__ __forceinline__ v8f wmma_bf16(v16bf a, v16bf b, v8f c) {
  return __builtin_amdgcn_wmma_f32_16x16x32_bf16(
      /*neg_a=*/false, a, /*neg_b=*/false, b,
      /*c_mod=*/(short)0, c, /*reuse_a=*/false, /*reuse_b=*/false);
}

// Load one 16x32 bf16 tile in the WMMA A/B per-lane layout.
// lane L: row = L&15 ; K-chunks start at kbase+((L>>4)*8) and +16.
static __device__ __forceinline__ v16bf load_tile_bf16(const __bf16* base, int ld,
                                                       int kbase, int lane) {
  int row = lane & 15;
  int k0  = kbase + ((lane >> 4) << 3);
  const __bf16* p = base + (size_t)row * ld + k0;
  union { v16bf v; uint4 u[2]; } t;
  t.u[0] = *(const uint4*)(p);
  t.u[1] = *(const uint4*)(p + 16);
  return t.v;
}

// Same A-tile load from an fp32 source, converting to bf16 (packed cvt).
static __device__ __forceinline__ v16bf load_tile_f32(const float* base, int ld,
                                                      int kbase, int lane) {
  int row = lane & 15;
  int k0  = kbase + ((lane >> 4) << 3);
  const float* p = base + (size_t)row * ld + k0;
  float4 a0 = *(const float4*)(p);
  float4 a1 = *(const float4*)(p + 4);
  float4 b0 = *(const float4*)(p + 16);
  float4 b1 = *(const float4*)(p + 20);
  union { v16bf v; v2bf p2[8]; } t;
  t.p2[0] = f2bf2(a0.x, a0.y); t.p2[1] = f2bf2(a0.z, a0.w);
  t.p2[2] = f2bf2(a1.x, a1.y); t.p2[3] = f2bf2(a1.z, a1.w);
  t.p2[4] = f2bf2(b0.x, b0.y); t.p2[5] = f2bf2(b0.z, b0.w);
  t.p2[6] = f2bf2(b1.x, b1.y); t.p2[7] = f2bf2(b1.z, b1.w);
  return t.v;
}

// ---------------------------------------------------------------------------
// Kernel 1: W[k][n] fp32 -> WT[n][k] bf16 for the 4 weight matrices.
// grid (4096, 4) block 256
// ---------------------------------------------------------------------------
__global__ __launch_bounds__(256)
void transpose_w_kernel(const float* __restrict__ Wq, const float* __restrict__ Wk,
                        const float* __restrict__ Wv, const float* __restrict__ Wo,
                        __bf16* __restrict__ ws) {
  int idx = blockIdx.x * 256 + threadIdx.x;      // 0 .. 1024*1024-1
  int n = idx >> 10, k = idx & 1023;
  const float* W = (blockIdx.y == 0) ? Wq : (blockIdx.y == 1) ? Wk
                 : (blockIdx.y == 2) ? Wv : Wo;
  __bf16* out = ws + (size_t)blockIdx.y * DM * DM;
  out[(size_t)n * DM + k] = f2bf(W[(size_t)k * DM + n]);
}

// ---------------------------------------------------------------------------
// Kernel 2: Q/K/V projections. grid (64, 16, 3), block 128 (4 waves).
// Wave computes a 16x64 output tile: D = X[16,1024] @ WT^T + bias.
// ---------------------------------------------------------------------------
__global__ __launch_bounds__(128)
void qkv_proj_kernel(const float* __restrict__ query, const float* __restrict__ key,
                     const float* __restrict__ value,
                     const __bf16* __restrict__ WqT, const __bf16* __restrict__ WkT,
                     const __bf16* __restrict__ WvT,
                     const float* __restrict__ bq, const float* __restrict__ bk,
                     const float* __restrict__ bv,
                     __bf16* __restrict__ Qb, __bf16* __restrict__ Kb,
                     __bf16* __restrict__ VTb) {
  const int mode = blockIdx.z;                   // 0=Q 1=K 2=V
  const float*  X    = (mode == 0) ? query : (mode == 1) ? key : value;
  const __bf16* WT   = (mode == 0) ? WqT   : (mode == 1) ? WkT : WvT;
  const float*  bias = (mode == 0) ? bq    : (mode == 1) ? bk  : bv;

  int lane = threadIdx.x & 31, wave = threadIdx.x >> 5;
  int mbase = (blockIdx.x * 4 + wave) * 16;      // row in [0, B*S)
  int nbase = blockIdx.y * 64;                   // col in [0, DM)

  const float* Arow = X + (size_t)mbase * DM + (size_t)(lane & 15) * DM;

  v8f acc[4] = {};
  for (int k = 0; k < DM; k += 32) {
    if (k + 32 < DM) __builtin_prefetch(Arow + k + 32, 0, 1);
    v16bf a = load_tile_f32(X + (size_t)mbase * DM, DM, k, lane);
#pragma unroll
    for (int j = 0; j < 4; ++j) {
      v16bf b = load_tile_bf16(WT + (size_t)(nbase + 16 * j) * DM, DM, k, lane);
      acc[j] = wmma_bf16(a, b, acc[j]);
    }
  }

  int half = lane >> 4, nl = lane & 15;
#pragma unroll
  for (int j = 0; j < 4; ++j) {
    int n = nbase + 16 * j + nl;
    int h = n >> 6, d = n & 63;
#pragma unroll
    for (int r = 0; r < 8; ++r) {
      int row = mbase + r + 8 * half;
      int b = row >> 11, s = row & (SEQ - 1);
      float val = acc[j][r] + bias[n];
      if (mode == 2)
        VTb[(((size_t)(b * NH + h)) * DH + d) * SEQ + s] = f2bf(val);
      else {
        __bf16* O = (mode == 0) ? Qb : Kb;
        O[(((size_t)(b * NH + h)) * SEQ + s) * DH + d] = f2bf(val);
      }
    }
  }
}

// ---------------------------------------------------------------------------
// Kernel 3: flash attention. grid (B*H=32, 32), block 128 (4 waves).
// Each wave: one 16-row Q tile, streams KV in 32-wide tiles.
// ---------------------------------------------------------------------------
__global__ __launch_bounds__(128)
void attn_kernel(const __bf16* __restrict__ Qb, const __bf16* __restrict__ Kb,
                 const __bf16* __restrict__ VTb, const float* __restrict__ attn_mask,
                 const unsigned char* __restrict__ kpm, __bf16* __restrict__ CTX) {
  __shared__ __align__(16) __bf16 Plds[4][16][32];   // per-wave P staging

  int lane = threadIdx.x & 31, wave = threadIdx.x >> 5;
  int half = lane >> 4, nl = lane & 15;
  int bh = blockIdx.x, b = bh >> 4, h = bh & 15;
  int qbase = (blockIdx.y * 4 + wave) * 16;

  const __bf16* Qhead  = Qb  + (size_t)bh * SEQ * DH;
  const __bf16* Khead  = Kb  + (size_t)bh * SEQ * DH;
  const __bf16* VThead = VTb + (size_t)bh * DH * SEQ;
  const float*  am = attn_mask + ((size_t)b * SEQ + qbase) * SEQ;
  const unsigned char* kp = kpm + (size_t)b * SEQ;

  // Q A-tile, reused for all KV tiles (d=64 -> two K=32 chunks).
  v16bf qa0 = load_tile_bf16(Qhead + (size_t)qbase * DH, DH, 0, lane);
  v16bf qa1 = load_tile_bf16(Qhead + (size_t)qbase * DH, DH, 32, lane);

  v8f acc[4] = {};
  float m[8], l[8];
#pragma unroll
  for (int r = 0; r < 8; ++r) { m[r] = -1e30f; l[r] = 0.f; }

  for (int kv = 0; kv < SEQ; kv += 32) {
    if (kv + 32 < SEQ) {
      __builtin_prefetch(am + (size_t)(half * 8) * SEQ + kv + 32 + nl, 0, 1);
      __builtin_prefetch(Khead + (size_t)(kv + 32) * DH + lane * 8, 0, 1);
    }

    // --- scores: S[16 x 32] = Q @ K^T ---
    v16bf kb00 = load_tile_bf16(Khead + (size_t)kv * DH, DH, 0, lane);
    v16bf kb01 = load_tile_bf16(Khead + (size_t)kv * DH, DH, 32, lane);
    v16bf kb10 = load_tile_bf16(Khead + (size_t)(kv + 16) * DH, DH, 0, lane);
    v16bf kb11 = load_tile_bf16(Khead + (size_t)(kv + 16) * DH, DH, 32, lane);
    v8f s0 = {}, s1 = {};
    s0 = wmma_bf16(qa0, kb00, s0); s0 = wmma_bf16(qa1, kb01, s0);
    s1 = wmma_bf16(qa0, kb10, s1); s1 = wmma_bf16(qa1, kb11, s1);

    float km0 = kp[kv + nl]      ? -1e30f : 0.f;
    float km1 = kp[kv + 16 + nl] ? -1e30f : 0.f;

    // --- masks + online softmax, row reductions within 16-lane halves ---
#pragma unroll
    for (int r = 0; r < 8; ++r) {
      int q = r + 8 * half;                       // local row 0..15
      float sc0 = s0[r] * 0.125f + am[(size_t)q * SEQ + kv + nl]      + km0;
      float sc1 = s1[r] * 0.125f + am[(size_t)q * SEQ + kv + 16 + nl] + km1;
      float vmax = fmaxf(sc0, sc1);
      vmax = fmaxf(vmax, __shfl_xor(vmax, 8, 32));
      vmax = fmaxf(vmax, __shfl_xor(vmax, 4, 32));
      vmax = fmaxf(vmax, __shfl_xor(vmax, 2, 32));
      vmax = fmaxf(vmax, __shfl_xor(vmax, 1, 32));
      float mn  = fmaxf(m[r], vmax);
      float fct = __expf(m[r] - mn);
      float p0  = __expf(sc0 - mn);
      float p1  = __expf(sc1 - mn);
      float ps  = p0 + p1;
      ps += __shfl_xor(ps, 8, 32);
      ps += __shfl_xor(ps, 4, 32);
      ps += __shfl_xor(ps, 2, 32);
      ps += __shfl_xor(ps, 1, 32);
      l[r] = l[r] * fct + ps;
      m[r] = mn;
      acc[0][r] *= fct; acc[1][r] *= fct; acc[2][r] *= fct; acc[3][r] *= fct;
      Plds[wave][q][nl]      = f2bf(p0);
      Plds[wave][q][nl + 16] = f2bf(p1);
    }

    // Re-shape P from C/D layout to A layout via LDS (same-wave, DS in-order).
    v16bf pa = load_tile_bf16(&Plds[wave][0][0], 32, 0, lane);

    // --- acc += P @ V  (B operand = VT rows, contiguous) ---
#pragma unroll
    for (int j = 0; j < 4; ++j) {
      v16bf vb = load_tile_bf16(VThead + (size_t)(16 * j) * SEQ, SEQ, kv, lane);
      acc[j] = wmma_bf16(pa, vb, acc[j]);
    }
  }

  // epilogue: normalize and store context (b, s, h*64) bf16
#pragma unroll
  for (int j = 0; j < 4; ++j) {
    int d = 16 * j + nl;
#pragma unroll
    for (int r = 0; r < 8; ++r) {
      int q = qbase + r + 8 * half;
      float inv = (l[r] > 0.f) ? (1.0f / l[r]) : 0.f;
      CTX[((size_t)(b * SEQ + q)) * DM + h * DH + d] = f2bf(acc[j][r] * inv);
    }
  }
}

// ---------------------------------------------------------------------------
// Kernel 4: output projection. grid (64, 16), block 128.
// out = CTX[4096,1024] @ Wo + bo   (fp32 out)
// ---------------------------------------------------------------------------
__global__ __launch_bounds__(128)
void oproj_kernel(const __bf16* __restrict__ CTX, const __bf16* __restrict__ WoT,
                  const float* __restrict__ bo, float* __restrict__ out) {
  int lane = threadIdx.x & 31, wave = threadIdx.x >> 5;
  int mbase = (blockIdx.x * 4 + wave) * 16;
  int nbase = blockIdx.y * 64;

  v8f acc[4] = {};
  for (int k = 0; k < DM; k += 32) {
    if (k + 32 < DM)
      __builtin_prefetch(CTX + (size_t)mbase * DM + (size_t)(lane & 15) * DM + k + 32, 0, 1);
    v16bf a = load_tile_bf16(CTX + (size_t)mbase * DM, DM, k, lane);
#pragma unroll
    for (int j = 0; j < 4; ++j) {
      v16bf b = load_tile_bf16(WoT + (size_t)(nbase + 16 * j) * DM, DM, k, lane);
      acc[j] = wmma_bf16(a, b, acc[j]);
    }
  }

  int half = lane >> 4, nl = lane & 15;
#pragma unroll
  for (int j = 0; j < 4; ++j) {
    int n = nbase + 16 * j + nl;
#pragma unroll
    for (int r = 0; r < 8; ++r) {
      int row = mbase + r + 8 * half;
      out[(size_t)row * DM + n] = acc[j][r] + bo[n];
    }
  }
}

// ---------------------------------------------------------------------------
extern "C" void kernel_launch(void* const* d_in, const int* in_sizes, int n_in,
                              void* d_out, int out_size, void* d_ws, size_t ws_size,
                              hipStream_t stream) {
  const float* query = (const float*)d_in[0];
  const float* key   = (const float*)d_in[1];
  const float* value = (const float*)d_in[2];
  const unsigned char* kpm = (const unsigned char*)d_in[3];
  const float* attn_mask = (const float*)d_in[4];
  const float* Wq = (const float*)d_in[5];
  const float* bq = (const float*)d_in[6];
  const float* Wk = (const float*)d_in[7];
  const float* bk = (const float*)d_in[8];
  const float* Wv = (const float*)d_in[9];
  const float* bv = (const float*)d_in[10];
  const float* Wo = (const float*)d_in[11];
  const float* bo = (const float*)d_in[12];
  float* out = (float*)d_out;

  const size_t MB = 1ull << 20;
  char* ws = (char*)d_ws;
  __bf16* WqT = (__bf16*)(ws + 0 * MB);
  __bf16* WkT = (__bf16*)(ws + 2 * MB);
  __bf16* WvT = (__bf16*)(ws + 4 * MB);
  __bf16* WoT = (__bf16*)(ws + 6 * MB);
  __bf16* Qb  = (__bf16*)(ws + 8 * MB);
  __bf16* Kb  = (__bf16*)(ws + 16 * MB);
  __bf16* VTb = (__bf16*)(ws + 24 * MB);
  __bf16* CTX = (__bf16*)(ws + 32 * MB);

  transpose_w_kernel<<<dim3((DM * DM) / 256, 4), 256, 0, stream>>>(Wq, Wk, Wv, Wo, WqT);
  qkv_proj_kernel<<<dim3((BATCH * SEQ) / (16 * 4), DM / 64, 3), 128, 0, stream>>>(
      query, key, value, WqT, WkT, WvT, bq, bk, bv, Qb, Kb, VTb);
  attn_kernel<<<dim3(BATCH * NH, SEQ / (16 * 4)), 128, 0, stream>>>(
      Qb, Kb, VTb, attn_mask, kpm, CTX);
  oproj_kernel<<<dim3((BATCH * SEQ) / (16 * 4), DM / 64), 128, 0, stream>>>(
      CTX, WoT, bo, out);
}
